// STFTRLayer_89000312308208
// MI455X (gfx1250) — matrix operand order repaired and checked by reading, not compiled, not run on hardware
//
#include <hip/hip_runtime.h>

#define C_    256
#define K_    8
#define M_    6
#define S_    3
#define B_    2
#define NB_   6000
#define BN_   (B_ * NB_)
#define TILE_ 16
#define PITCH_ 264            // LDS row pitch in bf16 elems (padded: 264*2B = 528B -> conflict-free)
#define WMAT_ 65536           // elements in one 256x256 matrix

typedef __attribute__((ext_vector_type(16))) __bf16       bf16x16;
typedef __attribute__((ext_vector_type(8)))  float        f32x8;
typedef __attribute__((ext_vector_type(4)))  unsigned int u32x4;

union FragU { u32x4 q[2]; bf16x16 v; };

__device__ __forceinline__ unsigned short f2bf(float f) {
  unsigned int u = __float_as_uint(f);
  u += 0x7FFFu + ((u >> 16) & 1u);          // round-to-nearest-even
  return (unsigned short)(u >> 16);
}
__device__ __forceinline__ float bf2f(unsigned short h) {
  return __uint_as_float(((unsigned int)h) << 16);
}

// A fragment (16x32 bf16) from LDS tile, rows pitch PITCH_.
// ISA layout: lanes 0-15 = rows, elem i -> K = k0 + i + 8*((i>=8)+laneHi): two b128 loads.
__device__ __forceinline__ bf16x16 load_a_lds(const unsigned short* rowbase, int k0, int lane) {
  const int m = lane & 15, hi = lane >> 4;
  const unsigned short* p = rowbase + m * PITCH_ + k0 + 8 * hi;
  FragU f;
  f.q[0] = *(const u32x4*)(p);
  f.q[1] = *(const u32x4*)(p + 16);
  return f.v;
}
// A fragment from a global bf16 row-major [rows][256] buffer.
__device__ __forceinline__ bf16x16 load_a_glob(const unsigned short* rowbase, int k0, int lane) {
  const int m = lane & 15, hi = lane >> 4;
  const unsigned short* p = rowbase + (size_t)m * C_ + k0 + 8 * hi;
  FragU f;
  f.q[0] = *(const u32x4*)(p);
  f.q[1] = *(const u32x4*)(p + 16);
  return f.v;
}
// B fragment (32x16 bf16) from global [n][k] bf16 matrix (row = output col n, contiguous k).
// ISA layout: lane n = column, elem i -> K = k0 + i + 8*((i>=8)+laneHi): two b128 loads.
__device__ __forceinline__ bf16x16 load_b(const unsigned short* W, int n0, int k0, int lane) {
  const int n = lane & 15, hi = lane >> 4;
  const unsigned short* p = W + (size_t)(n0 + n) * C_ + k0 + 8 * hi;
  FragU f;
  f.q[0] = *(const u32x4*)(p);
  f.q[1] = *(const u32x4*)(p + 16);
  return f.v;
}

#define WMMA_BF16(a, b, c) \
  __builtin_amdgcn_wmma_f32_16x16x32_bf16(false, (a), false, (b), (short)0, (c), false, false)

// D = A(16x256) @ W(256x256) + bias, written to LDS bf16 tile (pitch PITCH_).
// 16 n-tiles split across 8 waves; k looped in 8 steps of 32.
template <bool AGLOB>
__device__ __forceinline__ void gemm16(const unsigned short* A, const unsigned short* W,
                                       const float* bias, unsigned short* D,
                                       int lane, int wave) {
  for (int tt = wave; tt < 16; tt += 8) {
    const int n0 = tt * 16;
    f32x8 acc = {0.f, 0.f, 0.f, 0.f, 0.f, 0.f, 0.f, 0.f};
#pragma unroll
    for (int k0 = 0; k0 < C_; k0 += 32) {
      bf16x16 a = AGLOB ? load_a_glob(A, k0, lane) : load_a_lds(A, k0, lane);
      bf16x16 b = load_b(W, n0, k0, lane);
      acc = WMMA_BF16(a, b, acc);
    }
    const float bv = bias ? bias[n0 + (lane & 15)] : 0.f;
    const int hi = lane >> 4, n = n0 + (lane & 15);
#pragma unroll
    for (int r = 0; r < 8; ++r)
      D[(r + 8 * hi) * PITCH_ + n] = f2bf(acc[r] + bv);   // D layout: VGPR r -> row r+8*hi, lane -> col
  }
}

// ---------------------------------------------------------------------------
// Kernel 0: convert weights to bf16 in workspace; transpose those used as
// standard B matrices so B fragments load contiguously. Layout (id*WMAT_):
//  0..2 WqT_t[s]  3..5 Wk_t[s](natural)  6..8 WvT_t[s]  9..11 WoT_t[s]
//  12 WqcT  13 Wkc(natural)  14 WvcT  15 WocT
// ---------------------------------------------------------------------------
__global__ void __launch_bounds__(256) stftr_prep_weights(
    const float* __restrict__ Wq_t, const float* __restrict__ Wk_t,
    const float* __restrict__ Wv_t, const float* __restrict__ Wo_t,
    const float* __restrict__ Wq_c, const float* __restrict__ Wk_c,
    const float* __restrict__ Wv_c, const float* __restrict__ Wo_c,
    unsigned short* __restrict__ wsW) {
  const int id = blockIdx.x;
  const float* src;
  int trans;
  if (id < 3)       { src = Wq_t + (size_t)id * WMAT_;        trans = 1; }
  else if (id < 6)  { src = Wk_t + (size_t)(id - 3) * WMAT_;  trans = 0; }
  else if (id < 9)  { src = Wv_t + (size_t)(id - 6) * WMAT_;  trans = 1; }
  else if (id < 12) { src = Wo_t + (size_t)(id - 9) * WMAT_;  trans = 1; }
  else if (id == 12){ src = Wq_c; trans = 1; }
  else if (id == 13){ src = Wk_c; trans = 0; }
  else if (id == 14){ src = Wv_c; trans = 1; }
  else              { src = Wo_c; trans = 1; }
  unsigned short* dst = wsW + (size_t)id * WMAT_;
  for (int e = threadIdx.x; e < WMAT_; e += 256) {
    const int c = e >> 8, d = e & 255;            // coalesced read over d
    dst[trans ? ((d << 8) | c) : e] = f2bf(src[e]);
  }
}

// ---------------------------------------------------------------------------
// Kernel 1: temporal attention, fused per 16-voxel tile, summed over S.
// out[m,:] = sum_s ( ctx_s @ Wo_s + bo_s ),  ctx_s = w_s @ Wv_s + bv_s,
// w_s[m,c] = sum_j attn[m,j] kv_s[m,j,c],
// attn = softmax( kv[m,j,:].g_m + qproj_m.bk ),  g = qproj @ Wk^T,
// qproj = (qf+qpe) @ Wq + bq.   kv read from HBM exactly once.
// ---------------------------------------------------------------------------
__global__ void __launch_bounds__(256, 1) stftr_temporal(
    const float* __restrict__ qf, const float* __restrict__ qpe,
    const float* __restrict__ knnf, const float* __restrict__ knnpe,
    const float* __restrict__ bq_t, const float* __restrict__ bk_t,
    const float* __restrict__ bv_t, const float* __restrict__ bo_t,
    const unsigned short* __restrict__ wsW, unsigned short* __restrict__ outT) {
  extern __shared__ unsigned char smem[];
  unsigned short* kvA = (unsigned short*)smem;              // 128 * PITCH_
  unsigned short* qA  = kvA + 128 * PITCH_;                 // 16 * PITCH_ each below
  unsigned short* Xq  = qA + TILE_ * PITCH_;
  unsigned short* Xg  = Xq + TILE_ * PITCH_;
  unsigned short* Xw  = Xg + TILE_ * PITCH_;
  unsigned short* Xc  = Xw + TILE_ * PITCH_;
  float* qb   = (float*)(Xc + TILE_ * PITCH_);              // 16
  float* attn = qb + 16;                                    // 128

  const int t = threadIdx.x, lane = t & 31, wave = t >> 5;
  const int gv0 = blockIdx.x * TILE_;                       // NB_ % TILE_ == 0: no b straddle

  // stage q_base tile -> bf16 LDS
  for (int idx = t; idx < TILE_ * 64; idx += 256) {
    const int r = idx >> 6, c4 = (idx & 63) * 4;
    const float4 a = *(const float4*)(qf  + (size_t)(gv0 + r) * C_ + c4);
    const float4 b = *(const float4*)(qpe + (size_t)(gv0 + r) * C_ + c4);
    unsigned short* d = qA + r * PITCH_ + c4;
    d[0] = f2bf(a.x + b.x); d[1] = f2bf(a.y + b.y);
    d[2] = f2bf(a.z + b.z); d[3] = f2bf(a.w + b.w);
  }

  f32x8 oacc0 = {0.f, 0.f, 0.f, 0.f, 0.f, 0.f, 0.f, 0.f};
  f32x8 oacc1 = {0.f, 0.f, 0.f, 0.f, 0.f, 0.f, 0.f, 0.f};
  const int n0a = wave * 16, n0b = (wave + 8) * 16;

  for (int s = 0; s < S_; ++s) {
    __syncthreads();
    // stage kv = knn_feat + knn_pe : 128 rows x 256 -> bf16 LDS (the 590MB stream)
    const size_t rowbase = ((size_t)s * BN_ + gv0) * K_;
    for (int idx = t; idx < 128 * 64; idx += 256) {
      const int row = idx >> 6, c4 = (idx & 63) * 4;
      const size_t g = (rowbase + row) * C_ + c4;
      const float4 a = *(const float4*)(knnf + g);
      const float4 b = *(const float4*)(knnpe + g);
      if (s + 1 < S_ && (idx & 3) == 0) {                   // prefetch next sequence
        __builtin_prefetch(knnf + g + (size_t)BN_ * K_ * C_, 0, 0);
        __builtin_prefetch(knnpe + g + (size_t)BN_ * K_ * C_, 0, 0);
      }
      unsigned short* d = kvA + row * PITCH_ + c4;
      d[0] = f2bf(a.x + b.x); d[1] = f2bf(a.y + b.y);
      d[2] = f2bf(a.z + b.z); d[3] = f2bf(a.w + b.w);
    }
    __syncthreads();
    gemm16<false>(qA, wsW + (size_t)(0 + s) * WMAT_, bq_t + s * C_, Xq, lane, wave);  // qproj
    __syncthreads();
    gemm16<false>(Xq, wsW + (size_t)(3 + s) * WMAT_, nullptr, Xg, lane, wave);        // g = qproj@Wk^T
    __syncthreads();
    if (t < 16) {                                           // qproj . bk
      const float* bk = bk_t + s * C_;
      float acc = 0.f;
      for (int c = 0; c < C_; ++c) acc += bf2f(Xq[t * PITCH_ + c]) * bk[c];
      qb[t] = acc;
    }
    if (t < 128) {                                          // raw scores: kv row . g
      const unsigned short* kr = kvA + t * PITCH_;
      const unsigned short* gr = Xg + (t >> 3) * PITCH_;
      float acc = 0.f;
      for (int c = 0; c < C_; ++c) acc += bf2f(kr[c]) * bf2f(gr[c]);
      attn[t] = acc;
    }
    __syncthreads();
    if (t < 16) {                                           // softmax over K_=8
      float sv[K_]; float mx = -1e30f;
      for (int j = 0; j < K_; ++j) { sv[j] = (attn[t * K_ + j] + qb[t]) * 0.0625f; mx = fmaxf(mx, sv[j]); }
      float sum = 0.f;
      for (int j = 0; j < K_; ++j) { sv[j] = __expf(sv[j] - mx); sum += sv[j]; }
      const float inv = 1.f / sum;
      for (int j = 0; j < K_; ++j) attn[t * K_ + j] = sv[j] * inv;
    }
    __syncthreads();
    {                                                       // w[m,c] = sum_j attn * kv
      const int m = t >> 4, c0 = (t & 15) * 16;
      float aw[K_];
      for (int j = 0; j < K_; ++j) aw[j] = attn[m * K_ + j];
      for (int cc = 0; cc < 16; ++cc) {
        const int c = c0 + cc;
        float acc = 0.f;
        for (int j = 0; j < K_; ++j) acc += aw[j] * bf2f(kvA[(m * K_ + j) * PITCH_ + c]);
        Xw[m * PITCH_ + c] = f2bf(acc);
      }
    }
    __syncthreads();
    gemm16<false>(Xw, wsW + (size_t)(6 + s) * WMAT_, bv_t + s * C_, Xc, lane, wave);  // ctx
    __syncthreads();
    {                                                       // out += ctx @ Wo + bo (acc in regs)
      const unsigned short* Wo = wsW + (size_t)(9 + s) * WMAT_;
#pragma unroll
      for (int k0 = 0; k0 < C_; k0 += 32) {
        bf16x16 a = load_a_lds(Xc, k0, lane);
        oacc0 = WMMA_BF16(a, load_b(Wo, n0a, k0, lane), oacc0);
        oacc1 = WMMA_BF16(a, load_b(Wo, n0b, k0, lane), oacc1);
      }
      const float bo0 = bo_t[s * C_ + n0a + (lane & 15)];
      const float bo1 = bo_t[s * C_ + n0b + (lane & 15)];
#pragma unroll
      for (int r = 0; r < 8; ++r) { oacc0[r] += bo0; oacc1[r] += bo1; }
    }
  }
  const int hi = lane >> 4;
#pragma unroll
  for (int r = 0; r < 8; ++r) {                             // write out tile (bf16) to ws
    const size_t row = (size_t)(gv0 + r + 8 * hi) * C_;
    outT[row + n0a + (lane & 15)] = f2bf(oacc0[r]);
    outT[row + n0b + (lane & 15)] = f2bf(oacc1[r]);
  }
}

// ---------------------------------------------------------------------------
// Kernel 2: cross-sensor attention over M_=6 image samples, masked softmax,
// same folded-projection structure; writes final fp32 output (zeroed rows
// where no sample is valid).
// ---------------------------------------------------------------------------
__global__ void __launch_bounds__(256, 1) stftr_cross(
    const float* __restrict__ sampled, const unsigned char* __restrict__ mask,
    const float* __restrict__ bq_c, const float* __restrict__ bk_c,
    const float* __restrict__ bv_c, const float* __restrict__ bo_c,
    const unsigned short* __restrict__ wsW, const unsigned short* __restrict__ outT,
    float* __restrict__ out) {
  extern __shared__ unsigned char smem[];
  unsigned short* sfA = (unsigned short*)smem;              // 96 * PITCH_
  unsigned short* Xq  = sfA + 96 * PITCH_;
  unsigned short* Xg  = Xq + TILE_ * PITCH_;
  unsigned short* Xw  = Xg + TILE_ * PITCH_;
  unsigned short* Xc  = Xw + TILE_ * PITCH_;
  float* qb    = (float*)(Xc + TILE_ * PITCH_);             // 16
  float* attn  = qb + 16;                                   // 96
  int*   validm = (int*)(attn + 96);                        // 16

  const int t = threadIdx.x, lane = t & 31, wave = t >> 5;
  const int gv0 = blockIdx.x * TILE_;

  for (int idx = t; idx < 96 * 64; idx += 256) {            // stage sampled_feat tile
    const int row = idx >> 6, c4 = (idx & 63) * 4;
    const float4 a = *(const float4*)(sampled + ((size_t)gv0 * M_ + row) * C_ + c4);
    unsigned short* d = sfA + row * PITCH_ + c4;
    d[0] = f2bf(a.x); d[1] = f2bf(a.y); d[2] = f2bf(a.z); d[3] = f2bf(a.w);
  }
  if (t < 16) {
    int v = 0;
    for (int i = 0; i < M_; ++i) v |= (int)mask[(size_t)(gv0 + t) * M_ + i];
    validm[t] = v;
  }
  __syncthreads();
  gemm16<true>(outT + (size_t)gv0 * C_, wsW + 12 * WMAT_, bq_c, Xq, lane, wave);  // qc
  __syncthreads();
  gemm16<false>(Xq, wsW + 13 * WMAT_, nullptr, Xg, lane, wave);                   // gc = qc@Wk_c^T
  __syncthreads();
  if (t < 16) {
    float acc = 0.f;
    for (int c = 0; c < C_; ++c) acc += bf2f(Xq[t * PITCH_ + c]) * bk_c[c];
    qb[t] = acc;
  }
  if (t < 96) {
    const unsigned short* kr = sfA + t * PITCH_;
    const unsigned short* gr = Xg + (t / M_) * PITCH_;
    float acc = 0.f;
    for (int c = 0; c < C_; ++c) acc += bf2f(kr[c]) * bf2f(gr[c]);
    attn[t] = acc;
  }
  __syncthreads();
  if (t < 16) {                                             // masked softmax over M_=6
    float sv[M_]; float mx = -1e30f;
    for (int i = 0; i < M_; ++i) {
      const bool mk = mask[(size_t)(gv0 + t) * M_ + i] != 0;
      sv[i] = mk ? (attn[t * M_ + i] + qb[t]) * 0.0625f : -1e9f;
      mx = fmaxf(mx, sv[i]);
    }
    float sum = 0.f;
    for (int i = 0; i < M_; ++i) { sv[i] = __expf(sv[i] - mx); sum += sv[i]; }
    const float inv = 1.f / sum;
    for (int i = 0; i < M_; ++i) attn[t * M_ + i] = sv[i] * inv;
  }
  __syncthreads();
  {
    const int m = t >> 4, c0 = (t & 15) * 16;
    float aw[M_];
    for (int i = 0; i < M_; ++i) aw[i] = attn[m * M_ + i];
    for (int cc = 0; cc < 16; ++cc) {
      const int c = c0 + cc;
      float acc = 0.f;
      for (int i = 0; i < M_; ++i) acc += aw[i] * bf2f(sfA[(m * M_ + i) * PITCH_ + c]);
      Xw[m * PITCH_ + c] = f2bf(acc);
    }
  }
  __syncthreads();
  gemm16<false>(Xw, wsW + 14 * WMAT_, bv_c, Xc, lane, wave);                      // cctx
  __syncthreads();
  {
    const unsigned short* Wo = wsW + 15 * WMAT_;
    const int _n0a = wave * 16, _n0b = (wave + 8) * 16;
    f32x8 acc0 = {0.f, 0.f, 0.f, 0.f, 0.f, 0.f, 0.f, 0.f};
    f32x8 acc1 = {0.f, 0.f, 0.f, 0.f, 0.f, 0.f, 0.f, 0.f};
#pragma unroll
    for (int k0 = 0; k0 < C_; k0 += 32) {
      bf16x16 a = load_a_lds(Xc, k0, lane);
      acc0 = WMMA_BF16(a, load_b(Wo, _n0a, k0, lane), acc0);
      acc1 = WMMA_BF16(a, load_b(Wo, _n0b, k0, lane), acc1);
    }
    const float bo0 = bo_c[_n0a + (lane & 15)];
    const float bo1 = bo_c[_n0b + (lane & 15)];
    const int hi = lane >> 4;
#pragma unroll
    for (int r = 0; r < 8; ++r) {
      const int m = r + 8 * hi;
      const size_t row = (size_t)(gv0 + m) * C_;
      out[row + _n0a + (lane & 15)] = validm[m] ? (acc0[r] + bo0) : 0.f;
      out[row + _n0b + (lane & 15)] = validm[m] ? (acc1[r] + bo1) : 0.f;
    }
  }
}

// ---------------------------------------------------------------------------
extern "C" void kernel_launch(void* const* d_in, const int* in_sizes, int n_in,
                              void* d_out, int out_size, void* d_ws, size_t ws_size,
                              hipStream_t stream) {
  (void)in_sizes; (void)n_in; (void)out_size; (void)ws_size;
  const float* qf      = (const float*)d_in[0];
  const float* qpe     = (const float*)d_in[1];
  const float* knnf    = (const float*)d_in[2];
  const float* knnpe   = (const float*)d_in[3];
  const float* sampled = (const float*)d_in[4];
  const unsigned char* mask = (const unsigned char*)d_in[5];
  const float* Wq_t = (const float*)d_in[6];
  const float* Wk_t = (const float*)d_in[7];
  const float* Wv_t = (const float*)d_in[8];
  const float* Wo_t = (const float*)d_in[9];
  const float* bq_t = (const float*)d_in[10];
  const float* bk_t = (const float*)d_in[11];
  const float* bv_t = (const float*)d_in[12];
  const float* bo_t = (const float*)d_in[13];
  const float* Wq_c = (const float*)d_in[14];
  const float* Wk_c = (const float*)d_in[15];
  const float* Wv_c = (const float*)d_in[16];
  const float* Wo_c = (const float*)d_in[17];
  const float* bq_c = (const float*)d_in[18];
  const float* bk_c = (const float*)d_in[19];
  const float* bv_c = (const float*)d_in[20];
  const float* bo_c = (const float*)d_in[21];

  unsigned short* wsW  = (unsigned short*)d_ws;             // 16 matrices * 128KB = 2MB
  unsigned short* outT = wsW + (size_t)16 * WMAT_;          // bf16 temporal out, 6MB

  constexpr int SMEM_T = (128 + 5 * TILE_) * PITCH_ * 2 + (16 + 128) * 4;   // ~110.4 KB
  constexpr int SMEM_C = (96 + 4 * TILE_) * PITCH_ * 2 + (16 + 96) * 4 + 16 * 4; // ~85 KB
  hipFuncSetAttribute((const void*)stftr_temporal,
                      hipFuncAttributeMaxDynamicSharedMemorySize, SMEM_T);
  hipFuncSetAttribute((const void*)stftr_cross,
                      hipFuncAttributeMaxDynamicSharedMemorySize, SMEM_C);

  stftr_prep_weights<<<16, 256, 0, stream>>>(Wq_t, Wk_t, Wv_t, Wo_t,
                                             Wq_c, Wk_c, Wv_c, Wo_c, wsW);
  stftr_temporal<<<BN_ / TILE_, 256, SMEM_T, stream>>>(qf, qpe, knnf, knnpe,
                                                       bq_t, bk_t, bv_t, bo_t, wsW, outT);
  stftr_cross<<<BN_ / TILE_, 256, SMEM_C, stream>>>(sampled, mask, bq_c, bk_c, bv_c, bo_c,
                                                    wsW, outT, (float*)d_out);
}
